// LocationAttention_48868137894237
// MI455X (gfx1250) — compile-verified, matrix operand order checked by reading, not run on hardware
//
#include <hip/hip_runtime.h>
#include <hip/hip_bf16.h>
#include <math.h>

// ---------------------------------------------------------------------------
// Location attention, fused for MI455X (gfx1250, wave32, WMMA bf16 + TDM).
//   key_att GEMM:   (B*T x Ck) x (Ck x A)  via v_wmma_f32_16x16x32_bf16
//   tile staging:   tensor_load_to_lds (TDM 2D descriptor, padded LDS rows)
//   loc_att:        folded conv  W2[a,k] = Uw@conv_w  (31-tap FMA in epilogue)
//   normalization:  context = (sum_t s*mem) / (sum_t s)  -> single memory pass
// ---------------------------------------------------------------------------

typedef __attribute__((ext_vector_type(16))) __bf16 v16bf;
typedef __attribute__((ext_vector_type(8)))  float  v8f;
typedef __attribute__((ext_vector_type(4)))  unsigned int v4u;
typedef __attribute__((ext_vector_type(8)))  int v8i;
typedef __attribute__((ext_vector_type(4)))  int v4i;

#define T_     2000
#define B_     64
#define CK     512
#define CQ     1024
#define A_     128
#define F_     32
#define KCONV  31
#define WT     16                    // time rows per workgroup tile (2000/16=125)
#define NTILE  (T_ / WT)             // 125, exact
#define TROW   520                   // fp32 LDS row stride (TDM pad: +4 dw / 256 dw)
#define CADJ(c) ((c) + 4 * ((c) >> 8))   // data col -> padded LDS col

// ---------------- prep: q_att[b,a] = query[b,:] . Ww[a,:] + Wb[a] ----------
__global__ void la_prep_qatt(const float* __restrict__ query,
                             const float* __restrict__ Ww,
                             const float* __restrict__ Wb,
                             float* __restrict__ qatt) {
  int idx = blockIdx.x * blockDim.x + threadIdx.x;   // 8192 threads
  int b = idx >> 7, a = idx & 127;
  const float4* q4 = (const float4*)(query + b * CQ);
  const float4* w4 = (const float4*)(Ww + a * CQ);
  float s = 0.f;
  for (int c = 0; c < CQ / 4; ++c) {
    float4 q = q4[c], w = w4[c];
    s += q.x * w.x + q.y * w.y + q.z * w.z + q.w * w.w;
  }
  qatt[idx] = s + Wb[a];
}

// ---------------- prep: W2[k,a] = sum_f Uw[a,f]*conv_w[f,k]; row 31 = b2 ---
__global__ void la_prep_w2(const float* __restrict__ Uw,
                           const float* __restrict__ conv_w,
                           const float* __restrict__ conv_b,
                           float* __restrict__ w2t) {
  int a = threadIdx.x;               // one block of 128
  for (int k = 0; k < KCONV; ++k) {
    float s = 0.f;
    for (int f = 0; f < F_; ++f) s += Uw[a * F_ + f] * conv_w[f * KCONV + k];
    w2t[k * A_ + a] = s;
  }
  float b2 = 0.f;
  for (int f = 0; f < F_; ++f) b2 += Uw[a * F_ + f] * conv_b[f];
  w2t[KCONV * A_ + a] = b2;
}

// ---------------- prep: Vw -> bf16 B-matrix WMMA fragments -----------------
// frag layout: [kt(16)][ntile(8)][lane(32)][half(16)] bf16, 32B per lane.
__global__ void la_prep_vw(const float* __restrict__ Vw,
                           __bf16* __restrict__ vwfrag) {
  int idx = blockIdx.x * blockDim.x + threadIdx.x;   // 65536 threads
  int h    = idx & 15;
  int lane = (idx >> 4) & 31;
  int nt   = (idx >> 9) & 7;
  int kt   = idx >> 12;
  int n, k;
  if (lane < 16) { n = lane;      k = (h < 8) ? h      : h + 8;  }
  else           { n = lane - 16; k = (h < 8) ? h + 8  : h + 16; }
  vwfrag[idx] = (__bf16)Vw[(nt * 16 + n) * CK + (kt * 32 + k)];
}

// ---------------- main fused kernel ----------------------------------------
__global__ void __launch_bounds__(128)
la_main(const float* __restrict__ memory,          // [T,B,Ck]
        const float* __restrict__ state,           // [B,1,T]
        const unsigned char* __restrict__ mask,    // [B,T]
        const float* __restrict__ qatt,            // [B,A]
        const float* __restrict__ w2t,             // [32,A] (row 31 = b2)
        const __bf16* __restrict__ vwfrag,
        const float* __restrict__ ww,              // [1,A]
        float* __restrict__ sbuf,                  // [B,T] sigmoid values
        float* __restrict__ spart,                 // [B,NTILE]
        float* __restrict__ ctxpart) {             // [B,NTILE,Ck]
  __shared__ __align__(16) float s_tile[WT * TROW];    // 33280 B fp32 mem tile
  __shared__ float s_state[WT + 32];                   // state window t0-15..t0+30
  __shared__ float s_W2[KCONV * A_];                   // 15872 B
  __shared__ float s_qb[A_];                           // qatt + b2
  __shared__ float s_ww[A_];
  __shared__ float s_epart[4][WT];
  __shared__ float s_s[WT];

  const int tid = threadIdx.x;
  const int tt = blockIdx.x, b = blockIdx.y;
  const int t0 = tt * WT;

  // ---- TDM: DMA the 16 x 512 fp32 tile into padded LDS rows ---------------
  if (tid < 32) {                                      // wave 0 issues the DMA
    unsigned lds_base = (unsigned)(uintptr_t)(void*)s_tile;
    unsigned long long ga =
        (unsigned long long)(uintptr_t)(memory + ((size_t)t0 * B_ + b) * CK);
    union { v4u v; unsigned w[4]; } g0;
    g0.w[0] = 1u;                                      // count=1 (valid D#)
    g0.w[1] = lds_base;                                // lds_addr (bytes)
    g0.w[2] = (unsigned)(ga & 0xFFFFFFFFu);            // global_addr[31:0]
    g0.w[3] = (unsigned)((ga >> 32) & 0x01FFFFFFu)     // global_addr[56:32]
            | (2u << 30);                              // type=2 ("image")
    union { v8i v; unsigned w[8]; } g1;
    g1.w[0] = (2u << 16)        // data_size = 4 bytes
            | (1u << 20)        // pad_enable
            | (7u << 22)        // pad_interval = 256 dwords
            | (3u << 25);       // pad_amount   = 4 dwords  -> row stride 520 f
    g1.w[1] = ((unsigned)CK) << 16;     // tensor_dim0[15:0] @ bits 63:48
    g1.w[2] = ((unsigned)WT) << 16;     // dim0 hi=0 | tensor_dim1[15:0] @ 95:80
    g1.w[3] = ((unsigned)CK) << 16;     // dim1 hi=0 | tile_dim0 @ 127:112
    g1.w[4] = (unsigned)WT;             // tile_dim1 @ 143:128, tile_dim2=0
    g1.w[5] = (unsigned)(B_ * CK);      // tensor_dim0_stride[31:0] @ 191:160
    g1.w[6] = 0u;                       // stride hi | tensor_dim1_stride lo
    g1.w[7] = 0u;
    union { v4i v; int w[4]; } gz;
    gz.w[0] = gz.w[1] = gz.w[2] = gz.w[3] = 0;
    union { v8i v; int w[8]; } gz8;
    for (int i = 0; i < 8; ++i) gz8.w[i] = 0;
    __builtin_amdgcn_tensor_load_to_lds(g0.v, g1.v, gz.v, gz.v, gz8.v, 0);
  }

  // ---- stage small operands (other waves overlap with the DMA) ------------
  for (int i = 0; i < KCONV; ++i) s_W2[i * A_ + tid] = w2t[i * A_ + tid];
  s_qb[tid] = qatt[b * A_ + tid] + w2t[KCONV * A_ + tid];
  s_ww[tid] = ww[tid];
  if (tid < WT + 30) {
    int t = t0 - (KCONV / 2) + tid;
    s_state[tid] = (t >= 0 && t < T_) ? state[b * T_ + t] : 0.f;
  }
  if (tid < 32) __builtin_amdgcn_s_wait_tensorcnt(0);  // tile complete
  __syncthreads();

  // ---- WMMA GEMM: D[16 x A] = memTile(->bf16) x VwT(bf16) -----------------
  // wave w: N-tiles {2w, 2w+1}; register double-buffered A/B fragments.
  const int wave = tid >> 5, lane = tid & 31;
  const int arow = lane & 15;
  const int koff = (lane < 16) ? 0 : 8;
  const v16bf* __restrict__ vwfB =
      (const v16bf*)vwfrag + (wave * 2) * 32 + lane;   // + kt*256 + ntl*32
  const float* __restrict__ rowp = &s_tile[arow * TROW];

  union F4 { float4 v; float e[4]; };
  union AF { v16bf v; __bf16 e[16]; };
  v8f acc0 = {}, acc1 = {};

  // preload kt = 0
  v16bf bcur0 = vwfB[0], bcur1 = vwfB[32];
  F4 c0a, c0b, c1a, c1b;
  {
    int cl = CADJ(koff), ch = CADJ(16 + koff);
    c0a.v = *(const float4*)&rowp[cl];  c0b.v = *(const float4*)&rowp[cl + 4];
    c1a.v = *(const float4*)&rowp[ch];  c1b.v = *(const float4*)&rowp[ch + 4];
  }

#pragma unroll 2
  for (int kt = 0; kt < 16; ++kt) {
    // issue next iteration's loads first (latency hidden behind WMMAs)
    const int ktn = (kt < 15) ? kt + 1 : kt;
    v16bf bnx0 = vwfB[ktn * 256];
    v16bf bnx1 = vwfB[ktn * 256 + 32];
    F4 n0a, n0b, n1a, n1b;
    {
      int kc = ktn * 32;
      int cl = CADJ(kc + koff), ch = CADJ(kc + 16 + koff);
      n0a.v = *(const float4*)&rowp[cl]; n0b.v = *(const float4*)&rowp[cl + 4];
      n1a.v = *(const float4*)&rowp[ch]; n1b.v = *(const float4*)&rowp[ch + 4];
    }
    if (kt < 15)
      __builtin_prefetch(&vwfB[(ktn + 1 < 16 ? ktn + 1 : 15) * 256], 0, 0);

    AF af;
#pragma unroll
    for (int i = 0; i < 4; ++i) {
      af.e[i]      = (__bf16)c0a.e[i];
      af.e[4 + i]  = (__bf16)c0b.e[i];
      af.e[8 + i]  = (__bf16)c1a.e[i];
      af.e[12 + i] = (__bf16)c1b.e[i];
    }
    acc0 = __builtin_amdgcn_wmma_f32_16x16x32_bf16(
        false, af.v, false, bcur0, (short)0, acc0, false, false);
    acc1 = __builtin_amdgcn_wmma_f32_16x16x32_bf16(
        false, af.v, false, bcur1, (short)0, acc1, false, false);

    bcur0 = bnx0; bcur1 = bnx1;
    c0a = n0a; c0b = n0b; c1a = n1a; c1b = n1b;
  }

  // ---- epilogue: g = key_att + q_att + b2 + conv(state)*W2; e = ww.tanh(g) --
  const int halfsel = lane >> 4;
  const int lm = lane & 15;
#pragma unroll
  for (int j = 0; j < 8; ++j) {
    int r = j + halfsel * 8;                 // tile row 0..15
    float esum = 0.f;
#pragma unroll
    for (int ntl = 0; ntl < 2; ++ntl) {
      int a = wave * 32 + ntl * 16 + lm;
      float g = (ntl ? acc1[j] : acc0[j]) + s_qb[a];
      float conv = 0.f;
      for (int k = 0; k < KCONV; ++k) conv += s_state[r + k] * s_W2[k * A_ + a];
      g += conv;
      esum += tanhf(g) * s_ww[a];
    }
    for (int off = 8; off; off >>= 1) esum += __shfl_xor(esum, off, 16);
    if (lm == 0) s_epart[wave][r] = esum;    // unique writer per (wave, r)
  }
  __syncthreads();

  // ---- sigmoid + per-tile sum (lanes of wave 0) -----------------------------
  if (tid < WT) {
    int t = t0 + tid;
    float e = s_epart[0][tid] + s_epart[1][tid] + s_epart[2][tid] + s_epart[3][tid];
    float s = mask[b * T_ + t] ? 0.f : 1.f / (1.f + __expf(-e));
    sbuf[b * T_ + t] = s;
    s_s[tid] = s;
    float sum = s;
    for (int off = 8; off; off >>= 1) sum += __shfl_xor(sum, off, 16);
    if (tid == 0) spart[b * NTILE + tt] = sum;
  }
  __syncthreads();

  // ---- context partial: ctx[c] = sum_r s[r] * mem[r,c] (fp32 LDS tile) -----
  const int c0 = tid * 4;
  const int c0l = CADJ(c0);
  float a0 = 0.f, a1 = 0.f, a2 = 0.f, a3 = 0.f;
  for (int r = 0; r < WT; ++r) {
    float sv = s_s[r];
    const float* mrow = &s_tile[r * TROW + c0l];
    a0 += sv * mrow[0]; a1 += sv * mrow[1];
    a2 += sv * mrow[2]; a3 += sv * mrow[3];
  }
  float* cp = ctxpart + ((size_t)b * NTILE + tt) * CK + c0;
  *(float4*)cp = make_float4(a0, a1, a2, a3);
}

// ---------------- finalize: normalize, prob, next_state, context ------------
__global__ void la_finalize(const float* __restrict__ sbuf,
                            const float* __restrict__ spart,
                            const float* __restrict__ ctxpart,
                            const float* __restrict__ state,
                            float* __restrict__ out) {
  int b = blockIdx.x, tid = threadIdx.x;
  __shared__ float ssb;
  if (tid < 32) {
    float v = 0.f;
    for (int i = tid; i < NTILE; i += 32) v += spart[b * NTILE + i];
    for (int off = 16; off; off >>= 1) v += __shfl_xor(v, off, 32);
    if (tid == 0) ssb = v;
  }
  __syncthreads();
  float inv = 1.f / ssb;
  for (int c = tid; c < CK; c += 256) {
    float v = 0.f;
    for (int tl = 0; tl < NTILE; ++tl) v += ctxpart[((size_t)b * NTILE + tl) * CK + c];
    out[b * CK + c] = v * inv;                               // context [1,B,Ck]
  }
  for (int t = tid; t < T_; t += 256) {
    float p = sbuf[b * T_ + t] * inv;
    out[B_ * CK + b * T_ + t] = p;                           // prob [B,1,T]
    out[B_ * CK + B_ * T_ + b * T_ + t] = p + state[b * T_ + t];  // next_state
  }
}

// ---------------------------------------------------------------------------
extern "C" void kernel_launch(void* const* d_in, const int* in_sizes, int n_in,
                              void* d_out, int out_size, void* d_ws, size_t ws_size,
                              hipStream_t stream) {
  const float* memory = (const float*)d_in[0];
  const float* query  = (const float*)d_in[1];
  const float* state  = (const float*)d_in[2];
  const unsigned char* mask = (const unsigned char*)d_in[3];
  const float* Ww     = (const float*)d_in[4];
  const float* Wb     = (const float*)d_in[5];
  const float* Vw     = (const float*)d_in[6];
  const float* Uw     = (const float*)d_in[7];
  const float* conv_w = (const float*)d_in[8];
  const float* conv_b = (const float*)d_in[9];
  const float* ww     = (const float*)d_in[10];

  char* ws = (char*)d_ws;
  float*  qatt   = (float*)(ws);                         // 8192 f
  float*  w2t    = (float*)(ws + 32768);                 // 4096 f
  __bf16* vwfrag = (__bf16*)(ws + 49152);                // 65536 bf16
  float*  sbuf   = (float*)(ws + 180224);                // B*T f
  float*  spart  = (float*)(ws + 692224);                // B*NTILE f
  float*  ctxpart= (float*)(ws + 724224);                // B*NTILE*CK f (~16MB)

  la_prep_qatt<<<64, 128, 0, stream>>>(query, Ww, Wb, qatt);
  la_prep_w2<<<1, 128, 0, stream>>>(Uw, conv_w, conv_b, w2t);
  la_prep_vw<<<256, 256, 0, stream>>>(Vw, vwfrag);

  dim3 g1(NTILE, B_);
  la_main<<<g1, 128, 0, stream>>>(memory, state, mask, qatt, w2t, vwfrag, ww,
                                  sbuf, spart, ctxpart);
  la_finalize<<<B_, 256, 0, stream>>>(sbuf, spart, ctxpart, state, (float*)d_out);
}
